// ESN_14413910245612
// MI455X (gfx1250) — matrix operand order repaired and checked by reading, not compiled, not run on hardware
//
#include <hip/hip_runtime.h>
#include <stdint.h>

#define S_DIM 2048   // reservoir states
#define O_DIM 64     // outputs
#define T_LEN 512    // sequence length
#define B_DIM 64     // batch

typedef __attribute__((ext_vector_type(16))) __bf16 v16bf;
typedef __attribute__((ext_vector_type(8)))  __bf16 v8bf;
typedef __attribute__((ext_vector_type(8)))  float  v8f;

union BfVec { v16bf v; v8bf h[2]; };

__device__ __forceinline__ unsigned short f32_to_bf16(float f) {
  union { float f; uint32_t u; } c; c.f = f;
  uint32_t u = c.u;
  u += 0x7FFFu + ((u >> 16) & 1u);   // round-to-nearest-even
  return (unsigned short)(u >> 16);
}

// ---------------- prep kernels ----------------

__global__ void cvt_bf16_kernel(const float* __restrict__ src,
                                unsigned short* __restrict__ dst, int n) {
  int i = blockIdx.x * blockDim.x + threadIdx.x;
  int stride = gridDim.x * blockDim.x;
  for (; i < n; i += stride) dst[i] = f32_to_bf16(src[i]);
}

__global__ void init_state_kernel(const float* __restrict__ h_init,
                                  float* __restrict__ h,
                                  unsigned short* __restrict__ act, int n) {
  int i = blockIdx.x * blockDim.x + threadIdx.x;
  int stride = gridDim.x * blockDim.x;
  for (; i < n; i += stride) {
    float v = h_init[i];
    h[i] = v;
    act[i] = f32_to_bf16(tanhf(v));
  }
}

// ---------------- WMMA helpers ----------------

// One 16x16 f32 tile of A[16xK](bf16,row-major) * B^T where B rows are given
// row-major with length K. aRow/bRow already include the lane's row offset.
// A layout (16-bit 16x32): lanes0-15: K{0..7,16..23}; lanes16-31: K{8..15,24..31}
// B layout (32x16): lane=N; lanes0-15: K{0..7,16..23}; lanes16-31: K{8..15,24..31}
__device__ __forceinline__ v8f mm16x16_bf16(const unsigned short* __restrict__ aRow,
                                            const unsigned short* __restrict__ bRow,
                                            int K, int lane) {
  const int hiHalf = lane >> 4;
  const int offA = hiHalf * 8;
  const int offB = hiHalf * 8;
  v8f acc = {};
#pragma unroll 4
  for (int k0 = 0; k0 < K; k0 += 32) {
    BfVec a, b;
    a.h[0] = *(const v8bf*)(aRow + k0 + offA);        // K offA..offA+7
    a.h[1] = *(const v8bf*)(aRow + k0 + offA + 16);   // K offA+16..offA+23
    b.h[0] = *(const v8bf*)(bRow + k0 + offB);        // K offB..offB+7
    b.h[1] = *(const v8bf*)(bRow + k0 + offB + 16);   // K offB+16..offB+23
    acc = __builtin_amdgcn_wmma_f32_16x16x32_bf16(
        /*neg_a=*/false, a.v, /*neg_b=*/false, b.v,
        /*c_mod=*/(short)0, acc, /*reuse_a=*/false, /*reuse_b=*/false);
  }
  return acc;
}

// err[b,o] = sum_k act[b,k]*wo[o,k] + bo[o] - x[b,o], one 16x16 tile per wave
__device__ __forceinline__ void err_tile(int e, int lane,
                                         const unsigned short* __restrict__ act,
                                         const unsigned short* __restrict__ wo,
                                         const float* __restrict__ bo,
                                         const float* __restrict__ xt,
                                         float* __restrict__ outt) {
  const int bt = e >> 2, ot = e & 3;
  const int b0 = bt * 16, o0 = ot * 16;
  const int laneN = lane & 15;
  const unsigned short* aRow = act + (b0 + laneN) * S_DIM;
  const unsigned short* bRow = wo + (o0 + laneN) * S_DIM;
  v8f acc = mm16x16_bf16(aRow, bRow, S_DIM, lane);
  const int mbase = (lane >> 4) * 8;
  const int go = o0 + laneN;
  const float bias = bo[go];
#pragma unroll
  for (int r = 0; r < 8; ++r) {
    int gb = b0 + mbase + r;
    int idx = gb * O_DIM + go;
    outt[idx] = acc[r] + bias - xt[idx];
  }
}

// ---------------- per-timestep kernel ----------------
// blocks 0..127 (4 waves each, 512 waves): state update tiles for step t
// blocks 128..131 (16 waves): error GEMM for step t-1 (fused; skipped at t==0)
__global__ __launch_bounds__(128) void esn_step_kernel(
    float* __restrict__ h,
    const unsigned short* __restrict__ act_in,   // tanh(h_{t-1}) bf16
    unsigned short* __restrict__ act_out,        // tanh(h_t) bf16
    const unsigned short* __restrict__ wr,       // [S,S] bf16 row-major
    const float* __restrict__ br,
    const unsigned short* __restrict__ wo,       // [O,S] bf16 row-major
    const float* __restrict__ bo,
    const float* __restrict__ x_prev,            // x_{t-1} slice or null
    float* __restrict__ out_prev) {              // err_{t-1} slice or null
  const float inv_tau = 0.1f, decay = 0.9f;
  const int wave = threadIdx.x >> 5;
  const int lane = threadIdx.x & 31;

  if (blockIdx.x >= 128) {                       // fused error path (t-1)
    if (out_prev == nullptr) return;
    int e = (blockIdx.x - 128) * 4 + wave;       // 0..15
    err_tile(e, lane, act_in, wo, bo, x_prev, out_prev);
    return;
  }

  const int w = blockIdx.x * 4 + wave;           // 0..511
  const int bt = w >> 7;                         // batch tile 0..3
  const int st = w & 127;                        // state tile 0..127
  const int b0 = bt * 16, s0 = st * 16;
  const int laneN = lane & 15;

  const unsigned short* aRow = act_in + (b0 + laneN) * S_DIM;
  const unsigned short* bRow = wr + (s0 + laneN) * S_DIM;
  v8f acc = mm16x16_bf16(aRow, bRow, S_DIM, lane);

  const int mbase = (lane >> 4) * 8;
  const int gs = s0 + laneN;
  const float bias = br[gs];
#pragma unroll
  for (int r = 0; r < 8; ++r) {
    int gb = b0 + mbase + r;
    int idx = gb * S_DIM + gs;
    float hn = decay * h[idx] + inv_tau * (acc[r] + bias);
    h[idx] = hn;
    act_out[idx] = f32_to_bf16(tanhf(hn));
  }
}

// tail: error GEMM for the final timestep
__global__ __launch_bounds__(128) void esn_err_kernel(
    const unsigned short* __restrict__ act,
    const unsigned short* __restrict__ wo,
    const float* __restrict__ bo,
    const float* __restrict__ xt,
    float* __restrict__ outt) {
  int e = blockIdx.x * 4 + (threadIdx.x >> 5);   // grid 4 x 4 waves = 16 tiles
  err_tile(e, threadIdx.x & 31, act, wo, bo, xt, outt);
}

// ---------------- host launch ----------------

extern "C" void kernel_launch(void* const* d_in, const int* in_sizes, int n_in,
                              void* d_out, int out_size, void* d_ws, size_t ws_size,
                              hipStream_t stream) {
  const float* x      = (const float*)d_in[0];  // [T,B,O]
  const float* h_init = (const float*)d_in[1];  // [B,S]
  const float* w_r    = (const float*)d_in[2];  // [S,S]
  const float* b_r    = (const float*)d_in[3];  // [S]
  const float* w_o    = (const float*)d_in[4];  // [O,S]
  const float* b_o    = (const float*)d_in[5];  // [O]
  float* out = (float*)d_out;                   // [T,B,O]

  char* ws = (char*)d_ws;
  unsigned short* wr_bf = (unsigned short*)(ws);                       // 8 MiB
  unsigned short* wo_bf = (unsigned short*)(ws + (8u << 20));          // 256 KiB
  float*          h     = (float*)(ws + (8u << 20) + (256u << 10));    // 512 KiB
  unsigned short* act0  = (unsigned short*)(ws + (8u << 20) + (768u << 10)); // 256 KiB
  unsigned short* act1  = act0 + B_DIM * S_DIM;                        // 256 KiB

  // One-time (per launch) precision lowering + state init. W_r (8 MB bf16)
  // stays resident in the 192 MB L2 for all 512 steps.
  cvt_bf16_kernel<<<2048, 256, 0, stream>>>(w_r, wr_bf, S_DIM * S_DIM);
  cvt_bf16_kernel<<<128, 256, 0, stream>>>(w_o, wo_bf, O_DIM * S_DIM);
  init_state_kernel<<<128, 256, 0, stream>>>(h_init, h, act0, B_DIM * S_DIM);

  unsigned short* bufs[2] = {act0, act1};
  for (int t = 0; t < T_LEN; ++t) {
    unsigned short* ain  = bufs[t & 1];
    unsigned short* aout = bufs[(t + 1) & 1];
    const float* xp = (t == 0) ? nullptr : x   + (size_t)(t - 1) * B_DIM * O_DIM;
    float*       op = (t == 0) ? nullptr : out + (size_t)(t - 1) * B_DIM * O_DIM;
    esn_step_kernel<<<132, 128, 0, stream>>>(h, ain, aout, wr_bf, b_r,
                                             wo_bf, b_o, xp, op);
  }
  // t = 511 activations ended up in bufs[0]
  esn_err_kernel<<<4, 128, 0, stream>>>(bufs[0], wo_bf, b_o,
      x   + (size_t)(T_LEN - 1) * B_DIM * O_DIM,
      out + (size_t)(T_LEN - 1) * B_DIM * O_DIM);
}
